// KNNAttention_63075889709085
// MI455X (gfx1250) — matrix operand order repaired and checked by reading, not compiled, not run on hardware
//
#include <hip/hip_runtime.h>

typedef __attribute__((ext_vector_type(16))) __bf16 v16bf;
typedef __attribute__((ext_vector_type(8)))  __bf16 bf16x8;
typedef __attribute__((ext_vector_type(8)))  float  v8f;
typedef __attribute__((ext_vector_type(4)))  float  fx4;

#define WMMA_BF16(A, B, C) \
  __builtin_amdgcn_wmma_f32_16x16x32_bf16(false, (A), false, (B), (short)0, (C), false, false)

constexpr int kHeads = 12;
constexpr int kHd    = 64;
constexpr int kSeq   = 2048;
constexpr int kHid   = 768;   // kHeads * kHd

// softmax in base-2 domain (native v_exp_f32); scale folded into Q, mask folded into WMMA C
#define LOG2E 1.4426950408889634f
#define SCALE2 (0.125f * LOG2E)

__device__ __forceinline__ float fast_exp2(float x) { return __builtin_amdgcn_exp2f(x); }

// lane <-> lane^16 swap as pure VALU (v_permlanex16_b32), no LDS round-trip
__device__ __forceinline__ float xor16(float v) {
  int u = __builtin_bit_cast(int, v);
  int r = __builtin_amdgcn_permlanex16(u, u, 0x76543210, 0xfedcba98, false, false);
  return __builtin_bit_cast(float, r);
}

__device__ __forceinline__ v16bf cat8(bf16x8 lo, bf16x8 hi) {
  return __builtin_shufflevector(lo, hi, 0,1,2,3,4,5,6,7,8,9,10,11,12,13,14,15);
}

// ---------------- pre-pass 1: fp32 -> bf16, same layout (for K) ----------------
__global__ __launch_bounds__(256)
void cvt_bf16_kernel(const float* __restrict__ in, __bf16* __restrict__ out) {
  const size_t i = ((size_t)blockIdx.x * 256 + threadIdx.x) * 8;
  fx4 a = *(const fx4*)(in + i);
  fx4 b = *(const fx4*)(in + i + 4);
  bf16x8 r;
#pragma unroll
  for (int j = 0; j < 4; ++j) { r[j] = (__bf16)a[j]; r[j + 4] = (__bf16)b[j]; }
  *(bf16x8*)(out + i) = r;
}

// ---------------- pre-pass 2: V (b,s,h,d) fp32 -> VT (b,h,d,s) bf16 ----------------
__global__ __launch_bounds__(256)
void transpose_v_kernel(const float* __restrict__ V, __bf16* __restrict__ VT) {
  __shared__ __bf16 tile[64][66];   // pad to 66 to de-conflict transposed reads
  const int s0 = blockIdx.x * 64;
  const int h  = blockIdx.y;
  const int b  = blockIdx.z;
  const int t  = threadIdx.x;
  {
    const int r = t >> 2;
    const int c = t & 3;
    const float* src = V + ((size_t)(b * kSeq + s0 + r)) * kHid + h * kHd + c * 16;
#pragma unroll
    for (int k = 0; k < 4; ++k) {
      fx4 v4 = *(const fx4*)(src + k * 4);
#pragma unroll
      for (int j = 0; j < 4; ++j) tile[r][c * 16 + k * 4 + j] = (__bf16)v4[j];
    }
  }
  __syncthreads();
  {
    const int d  = t >> 2;
    const int sc = t & 3;
    __bf16* dst = VT + ((size_t)((b * kHeads + h) * kHd + d)) * kSeq + s0 + sc * 16;
    bf16x8 o0, o1;
#pragma unroll
    for (int j = 0; j < 8; ++j) {
      o0[j] = tile[sc * 16 + j][d];
      o1[j] = tile[sc * 16 + 8 + j][d];
    }
    *(bf16x8*)(dst)     = o0;
    *(bf16x8*)(dst + 8) = o1;
  }
}

// ---------------- pre-pass 3: mask * log2(e) ----------------
__global__ __launch_bounds__(256)
void scale_mask_kernel(const float* __restrict__ in, float* __restrict__ out) {
  const int i = blockIdx.x * 256 + threadIdx.x;
  out[i] = in[i] * LOG2E;
}

// ---------------- main flash-attention kernel ----------------
// One wave = 16 queries of one (b,h). Block = 4 waves = 64 queries.
// Score tiles computed transposed (keys in VGPRs/halves, queries on lanes);
// A-row permutation makes the P^T B-fragment purely lane-local (no shuffles);
// pre-scaled mask row is fed in as the QK^T WMMA C-operand.
__global__ __launch_bounds__(128)
void fa_fwd_kernel(const float* __restrict__ Q, const __bf16* __restrict__ Kb,
                   const __bf16* __restrict__ VT, const float* __restrict__ Mask2,
                   float* __restrict__ Out)
{
  const int lane = threadIdx.x & 31;
  const int wave = threadIdx.x >> 5;
  const int hi   = lane >> 4;
  const int ln   = lane & 15;
  const int lnp  = ln + 8 * (ln >> 3);   // permuted key-row offset: rows {0..7,16..23}
  const int h    = blockIdx.y;
  const int b    = blockIdx.z;
  const int q0   = blockIdx.x * 64 + wave * 16;

  const size_t bh_off = (size_t)b * kSeq * kHid + (size_t)h * kHd;

  // Resident Q^T B-fragments, pre-scaled by 0.125*log2e: elem i -> d = 32c + 8*hi + (i<8 ? i : i+8)
  v16bf Bq0, Bq1;
  {
    const float* qrow = Q + bh_off + (size_t)(q0 + ln) * kHid + 8 * hi;
#pragma unroll
    for (int c = 0; c < 2; ++c) {
      fx4 a = *(const fx4*)(qrow + 32 * c);
      fx4 bq = *(const fx4*)(qrow + 32 * c + 4);
      fx4 cc = *(const fx4*)(qrow + 32 * c + 16);
      fx4 dd = *(const fx4*)(qrow + 32 * c + 20);
      v16bf r;
#pragma unroll
      for (int i = 0; i < 4; ++i) {
        r[i]      = (__bf16)(a[i]  * SCALE2);
        r[i + 4]  = (__bf16)(bq[i] * SCALE2);
        r[i + 8]  = (__bf16)(cc[i] * SCALE2);
        r[i + 12] = (__bf16)(dd[i] * SCALE2);
      }
      if (c == 0) Bq0 = r; else Bq1 = r;
    }
  }

  v8f O0 = {}, O1 = {}, O2 = {}, O3 = {};     // O^T accumulators, d-chunks of 16
  float mrun = -1e30f;
  float lrun = 0.0f;

  const float*  m2row  = Mask2 + (size_t)b * kSeq;
  const __bf16* kbase  = Kb + bh_off;
  const __bf16* vtbase = VT + ((size_t)(b * kHeads + h) * kHd + ln) * kSeq;

  for (int n0 = 0; n0 < kSeq; n0 += 32) {
    // ---- mask row (already *log2e) becomes the accumulator init:
    //      E0[j] <-> key n0+16*hi+j ; E1[j] <-> key n0+16*hi+8+j
    fx4 ma = *(const fx4*)(m2row + n0 + 16 * hi);
    fx4 mb = *(const fx4*)(m2row + n0 + 16 * hi + 4);
    fx4 mc = *(const fx4*)(m2row + n0 + 16 * hi + 8);
    fx4 md = *(const fx4*)(m2row + n0 + 16 * hi + 12);
    v8f E0, E1;
#pragma unroll
    for (int j = 0; j < 4; ++j) {
      E0[j] = ma[j]; E0[j + 4] = mb[j];
      E1[j] = mc[j]; E1[j + 4] = md[j];
    }

    // ---- E^T = K_tile . Q' + mask : subtile0 rows = keys {0..7,16..23},
    //      subtile1 rows = keys {8..15,24..31} (lnp permutation)
    {
      const __bf16* krow0 = kbase + (size_t)(n0 + lnp) * kHid + 8 * hi;
      const __bf16* krow1 = krow0 + (size_t)8 * kHid;
      v16bf Ak;
      Ak = cat8(*(const bf16x8*)(krow0 +  0), *(const bf16x8*)(krow0 + 16));
      E0 = WMMA_BF16(Ak, Bq0, E0);
      Ak = cat8(*(const bf16x8*)(krow0 + 32), *(const bf16x8*)(krow0 + 48));
      E0 = WMMA_BF16(Ak, Bq1, E0);
      Ak = cat8(*(const bf16x8*)(krow1 +  0), *(const bf16x8*)(krow1 + 16));
      E1 = WMMA_BF16(Ak, Bq0, E1);
      Ak = cat8(*(const bf16x8*)(krow1 + 32), *(const bf16x8*)(krow1 + 48));
      E1 = WMMA_BF16(Ak, Bq1, E1);
    }

    // ---- online softmax (per query = per lane; keys 0..15 on half0, 16..31 on half1)
    float mloc = -1e30f;
#pragma unroll
    for (int j = 0; j < 8; ++j) mloc = fmaxf(mloc, fmaxf(E0[j], E1[j]));
    mloc = fmaxf(mloc, xor16(mloc));
    const float mnew  = fmaxf(mrun, mloc);
    const float alpha = fast_exp2(mrun - mnew);

    float p0[8], p1[8];
    float lloc = 0.0f;
#pragma unroll
    for (int j = 0; j < 8; ++j) {
      p0[j] = fast_exp2(E0[j] - mnew);
      p1[j] = fast_exp2(E1[j] - mnew);
      lloc += p0[j] + p1[j];
    }
    lloc += xor16(lloc);
    lrun = lrun * alpha + lloc;
    mrun = mnew;

#pragma unroll
    for (int j = 0; j < 8; ++j) {
      O0[j] *= alpha; O1[j] *= alpha; O2[j] *= alpha; O3[j] *= alpha;
    }

    // ---- P^T B-fragment: elem i -> key n0+16*hi+i : purely lane-local
    v16bf Bp;
#pragma unroll
    for (int i = 0; i < 8; ++i) {
      Bp[i]     = (__bf16)p0[i];
      Bp[i + 8] = (__bf16)p1[i];
    }

    // ---- O^T += V^T . P^T : A = V^T chunk (16 d-rows x 32 keys) from VT copy
#pragma unroll
    for (int dc = 0; dc < 4; ++dc) {
      const __bf16* vp = vtbase + (size_t)(dc * 16) * kSeq + n0 + 16 * hi;
      v16bf Av = cat8(*(const bf16x8*)(vp), *(const bf16x8*)(vp + 8));
      if      (dc == 0) O0 = WMMA_BF16(Av, Bp, O0);
      else if (dc == 1) O1 = WMMA_BF16(Av, Bp, O1);
      else if (dc == 2) O2 = WMMA_BF16(Av, Bp, O2);
      else              O3 = WMMA_BF16(Av, Bp, O3);
    }
  }

  // ---- epilogue: O^T[d][q] -> Out[b][q][h*64 + d]
  const float invl = 1.0f / lrun;
  float* orow = Out + bh_off + (size_t)(q0 + ln) * kHid;
#pragma unroll
  for (int j = 0; j < 8; ++j) {
    orow[ 0 + 8 * hi + j] = O0[j] * invl;
    orow[16 + 8 * hi + j] = O1[j] * invl;
    orow[32 + 8 * hi + j] = O2[j] * invl;
    orow[48 + 8 * hi + j] = O3[j] * invl;
  }
}

// ---------------- fallback (no workspace): direct-fp32 kernel ----------------
__global__ __launch_bounds__(128)
void fa_fwd_fallback(const float* __restrict__ Q, const float* __restrict__ K,
                     const float* __restrict__ V, const float* __restrict__ Mask,
                     float* __restrict__ Out)
{
  const int lane = threadIdx.x & 31;
  const int wave = threadIdx.x >> 5;
  const int hi   = lane >> 4;
  const int ln   = lane & 15;
  const int h    = blockIdx.y;
  const int b    = blockIdx.z;
  const int q0   = blockIdx.x * 64 + wave * 16;
  const size_t bh_off = (size_t)b * kSeq * kHid + (size_t)h * kHd;

  v16bf Bq0, Bq1;
  {
    const float* qrow = Q + bh_off + (size_t)(q0 + ln) * kHid + 8 * hi;
#pragma unroll
    for (int c = 0; c < 2; ++c) {
      fx4 a = *(const fx4*)(qrow + 32 * c);
      fx4 bq = *(const fx4*)(qrow + 32 * c + 4);
      fx4 cc = *(const fx4*)(qrow + 32 * c + 16);
      fx4 dd = *(const fx4*)(qrow + 32 * c + 20);
      v16bf r;
#pragma unroll
      for (int i = 0; i < 4; ++i) {
        r[i] = (__bf16)a[i]; r[i + 4] = (__bf16)bq[i];
        r[i + 8] = (__bf16)cc[i]; r[i + 12] = (__bf16)dd[i];
      }
      if (c == 0) Bq0 = r; else Bq1 = r;
    }
  }

  v8f O0 = {}, O1 = {}, O2 = {}, O3 = {};
  float mrun = -1e30f, lrun = 0.0f;
  const float* mrow  = Mask + (size_t)b * kSeq;
  const float* kbase = K + bh_off;
  const float* vbase = V + bh_off + ln;

  for (int n0 = 0; n0 < kSeq; n0 += 32) {
    v8f E0 = {}, E1 = {};
    {
      const float* krow0 = kbase + (size_t)(n0 + ln) * kHid + 8 * hi;
      const float* krow1 = krow0 + (size_t)16 * kHid;
#pragma unroll
      for (int t = 0; t < 2; ++t) {
        const float* kr = t ? krow1 : krow0;
#pragma unroll
        for (int c = 0; c < 2; ++c) {
          fx4 a = *(const fx4*)(kr + 32 * c);
          fx4 bb = *(const fx4*)(kr + 32 * c + 4);
          fx4 cc = *(const fx4*)(kr + 32 * c + 16);
          fx4 dd = *(const fx4*)(kr + 32 * c + 20);
          v16bf Ak;
#pragma unroll
          for (int i = 0; i < 4; ++i) {
            Ak[i] = (__bf16)a[i]; Ak[i + 4] = (__bf16)bb[i];
            Ak[i + 8] = (__bf16)cc[i]; Ak[i + 12] = (__bf16)dd[i];
          }
          if (t == 0) E0 = WMMA_BF16(Ak, c ? Bq1 : Bq0, E0);
          else        E1 = WMMA_BF16(Ak, c ? Bq1 : Bq0, E1);
        }
      }
    }
    fx4 ma = *(const fx4*)(mrow + n0 + 8 * hi);
    fx4 mb = *(const fx4*)(mrow + n0 + 8 * hi + 4);
    fx4 mc = *(const fx4*)(mrow + n0 + 16 + 8 * hi);
    fx4 md = *(const fx4*)(mrow + n0 + 16 + 8 * hi + 4);
#pragma unroll
    for (int j = 0; j < 4; ++j) {
      E0[j]     = (ma[j] == 0.0f) ? E0[j]     * SCALE2 : ma[j] * LOG2E;
      E0[j + 4] = (mb[j] == 0.0f) ? E0[j + 4] * SCALE2 : mb[j] * LOG2E;
      E1[j]     = (mc[j] == 0.0f) ? E1[j]     * SCALE2 : mc[j] * LOG2E;
      E1[j + 4] = (md[j] == 0.0f) ? E1[j + 4] * SCALE2 : md[j] * LOG2E;
    }
    float mloc = -1e30f;
#pragma unroll
    for (int j = 0; j < 8; ++j) mloc = fmaxf(mloc, fmaxf(E0[j], E1[j]));
    mloc = fmaxf(mloc, xor16(mloc));
    const float mnew  = fmaxf(mrun, mloc);
    const float alpha = fast_exp2(mrun - mnew);
    float p0[8], p1[8];
    float lloc = 0.0f;
#pragma unroll
    for (int j = 0; j < 8; ++j) {
      p0[j] = fast_exp2(E0[j] - mnew);
      p1[j] = fast_exp2(E1[j] - mnew);
      lloc += p0[j] + p1[j];
    }
    lloc += xor16(lloc);
    lrun = lrun * alpha + lloc;
    mrun = mnew;
#pragma unroll
    for (int j = 0; j < 8; ++j) { O0[j] *= alpha; O1[j] *= alpha; O2[j] *= alpha; O3[j] *= alpha; }

    v16bf Bp;
#pragma unroll
    for (int i = 0; i < 8; ++i) {
      float o0 = xor16(p0[i]);
      float o1 = xor16(p1[i]);
      Bp[i]     = (__bf16)(hi ? o1 : p0[i]);
      Bp[i + 8] = (__bf16)(hi ? p1[i] : o0);
    }
#pragma unroll
    for (int dc = 0; dc < 4; ++dc) {
      const float* vp = vbase + (size_t)n0 * kHid + dc * 16;
      v16bf Av;
#pragma unroll
      for (int i = 0; i < 16; ++i) {
        const int key = ((i < 8) ? i : (i + 8)) + 8 * hi;
        Av[i] = (__bf16)vp[(size_t)key * kHid];
      }
      if      (dc == 0) O0 = WMMA_BF16(Av, Bp, O0);
      else if (dc == 1) O1 = WMMA_BF16(Av, Bp, O1);
      else if (dc == 2) O2 = WMMA_BF16(Av, Bp, O2);
      else              O3 = WMMA_BF16(Av, Bp, O3);
    }
  }
  const float invl = 1.0f / lrun;
  float* orow = Out + bh_off + (size_t)(q0 + ln) * kHid;
#pragma unroll
  for (int j = 0; j < 8; ++j) {
    orow[ 0 + 8 * hi + j] = O0[j] * invl;
    orow[16 + 8 * hi + j] = O1[j] * invl;
    orow[32 + 8 * hi + j] = O2[j] * invl;
    orow[48 + 8 * hi + j] = O3[j] * invl;
  }
}

extern "C" void kernel_launch(void* const* d_in, const int* in_sizes, int n_in,
                              void* d_out, int out_size, void* d_ws, size_t ws_size,
                              hipStream_t stream) {
  (void)n_in; (void)out_size;
  const float* Q    = (const float*)d_in[0];
  const float* K    = (const float*)d_in[1];
  const float* V    = (const float*)d_in[2];
  const float* Mask = (const float*)d_in[3];
  float* Out = (float*)d_out;

  const int B = in_sizes[0] / (kSeq * kHid);               // = 4
  const size_t elems = (size_t)B * kSeq * kHid;            // 6.29M
  const size_t mask_elems = (size_t)B * kSeq;              // 8192
  const size_t need  = elems * 2u * sizeof(__bf16) + mask_elems * sizeof(float);

  dim3 grid(kSeq / 64, kHeads, B);
  if (ws_size >= need) {
    __bf16* wsK  = (__bf16*)d_ws;
    __bf16* wsVT = wsK + elems;
    float*  wsM2 = (float*)(wsVT + elems);
    cvt_bf16_kernel<<<dim3((unsigned)(elems / 2048)), 256, 0, stream>>>(K, wsK);
    transpose_v_kernel<<<dim3(kSeq / 64, kHeads, B), 256, 0, stream>>>(V, wsVT);
    scale_mask_kernel<<<dim3((unsigned)(mask_elems / 256)), 256, 0, stream>>>(Mask, wsM2);
    fa_fwd_kernel<<<grid, 128, 0, stream>>>(Q, wsK, wsVT, wsM2, Out);
  } else {
    fa_fwd_fallback<<<grid, 128, 0, stream>>>(Q, K, V, Mask, Out);
  }
}
